// PTFactorNoisePredictor_73280732004885
// MI455X (gfx1250) — matrix operand order repaired and checked by reading, not compiled, not run on hardware
//
#include <hip/hip_runtime.h>
#include <math.h>

// ---------------------------------------------------------------------------
// Problem constants (from reference)
// ---------------------------------------------------------------------------
#define BB 8
#define CC 128
#define PP 96
#define DD 128
#define HH 8
#define RR 16
#define GG 512
#define NROWS (BB * CC * PP)          // 98304 rows of length D
#define NELEM ((size_t)NROWS * DD)    // 12,582,912

typedef __attribute__((ext_vector_type(16))) __bf16 v16bf;
typedef __attribute__((ext_vector_type(8)))  float  v8f;
typedef __attribute__((ext_vector_type(4)))  unsigned int v4u;
typedef __attribute__((ext_vector_type(8)))  int v8i;
typedef __attribute__((ext_vector_type(4)))  int v4i;

union BFrag { v16bf v; unsigned short u[16]; };

// round-to-nearest-even f32 -> bf16 (as raw u16)
static __device__ inline unsigned short f2bf(float f) {
    unsigned int x = __float_as_uint(f);
    unsigned int r = x + 0x7FFFu + ((x >> 16) & 1u);
    return (unsigned short)(r >> 16);
}

static __device__ inline v8f wmma_bf(const BFrag& a, const BFrag& b, v8f c) {
    return __builtin_amdgcn_wmma_f32_16x16x32_bf16(
        /*neg_a=*/false, a.v, /*neg_b=*/false, b.v,
        /*c_mod=*/(short)0, c, /*reuse_a=*/false, /*reuse_b=*/false);
}

// ---------------------------------------------------------------------------
// Tensor Data Mover: async 2-D bf16 tile copy global -> LDS.
// D# per CDNA5 ISA (data_size=2B, 2-D tensor,
// groups 2/3 zero).  Issue from ONE wave; TDM ignores EXEC.
// LDS byte address = low 32 bits of the flat pointer (AS3->flat lowering).
// clang-23 toolchain: 6-arg builtin (g0, g1, g2, g3, g4, cpol).
// ---------------------------------------------------------------------------
static __device__ inline void tdm_load_bf16_2d(const unsigned short* gsrc,
                                               const unsigned short* lds_dst,
                                               unsigned rowElems, unsigned nRows,
                                               unsigned strideElems) {
    unsigned long long ga = (unsigned long long)(const void*)gsrc;
    unsigned lds_off = (unsigned)(unsigned long long)(const void*)lds_dst;
    v4u g0;
    g0[0] = 1u;                                                  // count=1 (valid user D#)
    g0[1] = lds_off;                                             // lds_addr (bytes)
    g0[2] = (unsigned)ga;                                        // global_addr[31:0]
    g0[3] = (unsigned)((ga >> 32) & 0x01FFFFFFu) | (2u << 30);   // addr[56:32] | type=2
    v8i g1;
    g1[0] = (int)(1u << 16);                                     // wg_mask=0, data_size=1 (2B)
    g1[1] = (int)((rowElems & 0xFFFFu) << 16);                   // tensor_dim0[15:0]
    g1[2] = (int)(((rowElems >> 16) & 0xFFFFu) |                 // tensor_dim0[31:16]
                  ((nRows & 0xFFFFu) << 16));                    // tensor_dim1[15:0]
    g1[3] = (int)(((nRows >> 16) & 0xFFFFu) |                    // tensor_dim1[31:16]
                  ((rowElems & 0xFFFFu) << 16));                 // tile_dim0
    g1[4] = (int)(nRows & 0xFFFFu);                              // tile_dim1 (tile_dim2=0)
    g1[5] = (int)strideElems;                                    // tensor_dim0_stride[31:0]
    g1[6] = 0;
    g1[7] = 0;
    v4i z4 = {};
    v8i z8 = {};
    __builtin_amdgcn_tensor_load_to_lds(g0, g1, z4, z4, z8, 0);
}

static __device__ inline void tdm_wait0() {
    __builtin_amdgcn_s_wait_tensorcnt(0);
}

// ---------------------------------------------------------------------------
// Fragment loaders (CDNA5 wave32 WMMA VGPR layouts, CDNA5 ISA)
// ---------------------------------------------------------------------------
static __device__ inline BFrag frag_a_bf(const unsigned short* S, int ld, int m0, int k0) {
    BFrag f;
    int lane = threadIdx.x & 31, M = lane & 15, half = lane >> 4;
    const unsigned short* row = S + (m0 + M) * ld + k0;
#pragma unroll
    for (int i = 0; i < 8; ++i) {
        int k = (half << 4) + (i << 1);
        f.u[2 * i]     = row[k];
        f.u[2 * i + 1] = row[k + 1];
    }
    return f;
}

static __device__ inline BFrag frag_a_bf_k16(const unsigned short* S, int ld, int m0, int k0) {
    BFrag f;
    int lane = threadIdx.x & 31, M = lane & 15, half = lane >> 4;
    const unsigned short* row = S + (m0 + M) * ld + k0;
#pragma unroll
    for (int i = 0; i < 8; ++i) {
        int k = i << 1;
        if (!half) { f.u[2 * i] = row[k]; f.u[2 * i + 1] = row[k + 1]; }
        else       { f.u[2 * i] = 0;      f.u[2 * i + 1] = 0;          }
    }
    return f;
}

static __device__ inline BFrag frag_a_f32(const float* S, int ld, int m0, int k0,
                                          const float* rowscale) {
    BFrag f;
    int lane = threadIdx.x & 31, M = lane & 15, half = lane >> 4;
    const float* row = S + (m0 + M) * ld + k0;
    float sc = rowscale ? rowscale[m0 + M] : 1.0f;
#pragma unroll
    for (int i = 0; i < 8; ++i) {
        int k = (half << 4) + (i << 1);
        f.u[2 * i]     = f2bf(row[k] * sc);
        f.u[2 * i + 1] = f2bf(row[k + 1] * sc);
    }
    return f;
}

// B[k][n] = W[n0+n][k0+k] (computes X @ W^T); contiguous per lane
static __device__ inline BFrag frag_bt_bf(const unsigned short* W, int ld, int n0, int k0) {
    BFrag f;
    int lane = threadIdx.x & 31, N = lane & 15, half = lane >> 4;
    const unsigned short* row = W + (n0 + N) * ld + k0 + (half << 4);
#pragma unroll
    for (int j = 0; j < 8; ++j) {
        f.u[2 * j]     = row[2 * j];
        f.u[2 * j + 1] = row[2 * j + 1];
    }
    return f;
}

static __device__ inline BFrag frag_bt_bf_k16(const unsigned short* W, int ld, int n0, int k0) {
    BFrag f;
    int lane = threadIdx.x & 31, N = lane & 15, half = lane >> 4;
    const unsigned short* row = W + (n0 + N) * ld + k0;
#pragma unroll
    for (int j = 0; j < 8; ++j) {
        f.u[2 * j]     = half ? (unsigned short)0 : row[2 * j];
        f.u[2 * j + 1] = half ? (unsigned short)0 : row[2 * j + 1];
    }
    return f;
}

// B[k][n] = S[k0+k][n0+n] (plain A @ B)
static __device__ inline BFrag frag_b_bf(const unsigned short* S, int ld, int k0, int n0) {
    BFrag f;
    int lane = threadIdx.x & 31, N = lane & 15, half = lane >> 4;
    const unsigned short* col = S + (k0 + (half << 4)) * ld + n0 + N;
#pragma unroll
    for (int j = 0; j < 8; ++j) {
        f.u[2 * j]     = col[(2 * j) * ld];
        f.u[2 * j + 1] = col[(2 * j + 1) * ld];
    }
    return f;
}

static __device__ inline void st_c_f32(float* Dst, int ld, int m0, int n0, v8f c, float scale) {
    int lane = threadIdx.x & 31, N = lane & 15, half = lane >> 4;
#pragma unroll
    for (int i = 0; i < 8; ++i) Dst[(m0 + i + (half << 3)) * ld + n0 + N] = c[i] * scale;
}

static __device__ inline void st_c_f32_relu(float* Dst, int ld, int m0, int n0, v8f c) {
    int lane = threadIdx.x & 31, N = lane & 15, half = lane >> 4;
#pragma unroll
    for (int i = 0; i < 8; ++i) Dst[(m0 + i + (half << 3)) * ld + n0 + N] = fmaxf(c[i], 0.0f);
}

static __device__ inline void st_c_bf16(unsigned short* Dst, int ld, int m0, int n0, v8f c) {
    int lane = threadIdx.x & 31, N = lane & 15, half = lane >> 4;
#pragma unroll
    for (int i = 0; i < 8; ++i) Dst[(m0 + i + (half << 3)) * ld + n0 + N] = f2bf(c[i]);
}

// ---------------------------------------------------------------------------
// Kernel 0: f32 -> bf16 conversion (weights prepass), grid-stride
// ---------------------------------------------------------------------------
__global__ void k_f32_to_bf16(const float* __restrict__ src, unsigned short* __restrict__ dst,
                              int n) {
    for (int i = blockIdx.x * blockDim.x + threadIdx.x; i < n; i += gridDim.x * blockDim.x)
        dst[i] = f2bf(src[i]);
}

// ---------------------------------------------------------------------------
// Kernel 1: layernorm rows of D=128, one wave per row; writes f32 + bf16
// ---------------------------------------------------------------------------
__global__ void k_layernorm(const float* __restrict__ x, const float* __restrict__ gw,
                            const float* __restrict__ bw, float* __restrict__ y,
                            unsigned short* __restrict__ ybf, int nrows) {
    int row = blockIdx.x * (blockDim.x >> 5) + (threadIdx.x >> 5);
    if (row >= nrows) return;
    int lane = threadIdx.x & 31;
    const float4* rx = (const float4*)(x + (size_t)row * DD);
    float4 v = rx[lane];
    float s = v.x + v.y + v.z + v.w;
#pragma unroll
    for (int o = 16; o > 0; o >>= 1) s += __shfl_xor(s, o, 32);
    float mu = s * (1.0f / 128.0f);
    float dx = v.x - mu, dy = v.y - mu, dz = v.z - mu, dw = v.w - mu;
    float q = dx * dx + dy * dy + dz * dz + dw * dw;
#pragma unroll
    for (int o = 16; o > 0; o >>= 1) q += __shfl_xor(q, o, 32);
    float rstd = rsqrtf(q * (1.0f / 128.0f) + 1e-5f);
    int d = lane * 4;
    float4 ov;
    ov.x = dx * rstd * gw[d + 0] + bw[d + 0];
    ov.y = dy * rstd * gw[d + 1] + bw[d + 1];
    ov.z = dz * rstd * gw[d + 2] + bw[d + 2];
    ov.w = dw * rstd * gw[d + 3] + bw[d + 3];
    ((float4*)(y + (size_t)row * DD))[lane] = ov;
    uint2 pk;
    pk.x = (unsigned)f2bf(ov.x) | ((unsigned)f2bf(ov.y) << 16);
    pk.y = (unsigned)f2bf(ov.z) | ((unsigned)f2bf(ov.w) << 16);
    ((uint2*)(ybf + (size_t)row * DD))[lane] = pk;
}

// ---------------------------------------------------------------------------
// Kernel 2: time messages, one WG per (b,c).  msg <- mt (first writer)
// LDS: Xt 24K | U 32K | V 32K | Qu 24K | Qv 24K | Sc 36K | Ps 36K = 208 KB
// ---------------------------------------------------------------------------
#define SM_TIME (24576 + 32768 + 32768 + 24576 + 24576 + 36864 + 36864)
__global__ void k_time(const float* __restrict__ qzn, const unsigned short* __restrict__ qznb,
                       const unsigned short* __restrict__ Ug, const unsigned short* __restrict__ Vg,
                       float* __restrict__ msg) {
    extern __shared__ char smem[];
    unsigned short* Xt = (unsigned short*)smem;                              // 96x128
    unsigned short* Ul = (unsigned short*)(smem + 24576);                    // 128x128
    unsigned short* Vl = (unsigned short*)(smem + 24576 + 32768);            // 128x128
    unsigned short* Qu = (unsigned short*)(smem + 24576 + 65536);            // 96x128
    unsigned short* Qv = (unsigned short*)(smem + 24576 + 65536 + 24576);    // 96x128
    float* Sc = (float*)(smem + 24576 + 65536 + 49152);                      // 96x96
    float* Ps = (float*)(smem + 24576 + 65536 + 49152 + 36864);              // 96x96

    int b = blockIdx.x >> 7, c = blockIdx.x & 127;
    int tid = threadIdx.x, nthr = blockDim.x;
    int wave = tid >> 5, nw = nthr >> 5;

    // async TDM staging (wave 0 issues; TDM ignores EXEC)
    if (wave == 0) {
        tdm_load_bf16_2d(qznb + (((size_t)b * CC + c) * PP) * DD, Xt, PP * DD, 1, PP * DD);
        tdm_load_bf16_2d(Ug + (size_t)b * DD * DD, Ul, DD * DD, 1, DD * DD);
        tdm_load_bf16_2d(Vg + (size_t)b * DD * DD, Vl, DD * DD, 1, DD * DD);
        tdm_wait0();
    }
    __syncthreads();

    // Qu = X @ U^T ; Qv = X @ V^T
    for (int job = wave; job < 2 * 6 * 8; job += nw) {
        int uv = job >= 48;
        int jj = job & 47;
        int m0 = (jj >> 3) << 4, n0 = (jj & 7) << 4;
        const unsigned short* W = uv ? Vl : Ul;
        v8f acc = {};
#pragma unroll
        for (int k0 = 0; k0 < DD; k0 += 32)
            acc = wmma_bf(frag_a_bf(Xt, DD, m0, k0), frag_bt_bf(W, DD, n0, k0), acc);
        st_c_bf16(uv ? Qv : Qu, DD, m0, n0, acc);
    }
    for (int i = tid; i < PP * PP; i += nthr) Ps[i] = 0.0f;
    __syncthreads();

    // per-head scores + softmax, accumulate probs/H into Ps
    for (int h = 0; h < HH; ++h) {
        int kh = h << 4;
        for (int job = wave; job < 6 * 6; job += nw) {
            int m0 = (job / 6) << 4, n0 = (job % 6) << 4;
            v8f acc = {};
            acc = wmma_bf(frag_a_bf_k16(Qu, DD, m0, kh), frag_bt_bf_k16(Qv, DD, n0, kh), acc);
            st_c_f32(Sc, PP, m0, n0, acc, 0.25f);  // 1/sqrt(R)
        }
        __syncthreads();
        for (int r = tid; r < PP; r += nthr) {
            const float* row = Sc + r * PP;
            float mx = -3.4e38f;
            for (int qq = 0; qq < PP; ++qq) mx = fmaxf(mx, row[qq]);
            float sum = 0.0f;
            for (int qq = 0; qq < PP; ++qq) sum += __expf(row[qq] - mx);
            float inv = 1.0f / (sum * (float)HH);
            float* prow = Ps + r * PP;
            for (int qq = 0; qq < PP; ++qq) prow[qq] += __expf(row[qq] - mx) * inv;
        }
        __syncthreads();
    }

    // mt = Ps @ X   (sum over q = 96)
    float* Mg = msg + (((size_t)b * CC + c) * PP) * DD;
    for (int job = wave; job < 6 * 8; job += nw) {
        int m0 = (job >> 3) << 4, n0 = (job & 7) << 4;
        v8f acc = {};
#pragma unroll
        for (int k0 = 0; k0 < PP; k0 += 32)
            acc = wmma_bf(frag_a_f32(Ps, PP, m0, k0, nullptr), frag_b_bf(Xt, DD, k0, n0), acc);
        int lane = threadIdx.x & 31, N = lane & 15, half = lane >> 4;
#pragma unroll
        for (int i = 0; i < 8; ++i) Mg[(m0 + i + (half << 3)) * DD + n0 + N] = acc[i];
    }
}

// ---------------------------------------------------------------------------
// Kernel 3: channel messages, one WG per (b,p).  msg += mc
// LDS: Xc 32K | U 32K | V 32K | Cu 32K | Cv 32K | Sc 64K | Ps 64K = 288 KB
// X tile is a strided row gather -> single 2-D TDM descriptor.
// ---------------------------------------------------------------------------
#define SM_CH (32768 * 5 + 65536 * 2)
__global__ void k_channel(const float* __restrict__ qzn, const unsigned short* __restrict__ qznb,
                          const unsigned short* __restrict__ Ug, const unsigned short* __restrict__ Vg,
                          float* __restrict__ msg) {
    extern __shared__ char smem[];
    unsigned short* Xc = (unsigned short*)smem;
    unsigned short* Ul = (unsigned short*)(smem + 32768);
    unsigned short* Vl = (unsigned short*)(smem + 65536);
    unsigned short* Cu = (unsigned short*)(smem + 98304);
    unsigned short* Cv = (unsigned short*)(smem + 131072);
    float* Sc = (float*)(smem + 163840);
    float* Ps = (float*)(smem + 229376);

    int b = blockIdx.x / PP, p = blockIdx.x % PP;
    int tid = threadIdx.x, nthr = blockDim.x;
    int wave = tid >> 5, nw = nthr >> 5;

    if (wave == 0) {
        // rows c = 0..127, row length D, row stride P*D elements
        tdm_load_bf16_2d(qznb + (((size_t)b * CC) * PP + p) * DD, Xc, DD, CC, PP * DD);
        tdm_load_bf16_2d(Ug + (size_t)b * DD * DD, Ul, DD * DD, 1, DD * DD);
        tdm_load_bf16_2d(Vg + (size_t)b * DD * DD, Vl, DD * DD, 1, DD * DD);
        tdm_wait0();
    }
    __syncthreads();

    for (int job = wave; job < 2 * 8 * 8; job += nw) {
        int uv = job >= 64;
        int jj = job & 63;
        int m0 = (jj >> 3) << 4, n0 = (jj & 7) << 4;
        const unsigned short* W = uv ? Vl : Ul;
        v8f acc = {};
#pragma unroll
        for (int k0 = 0; k0 < DD; k0 += 32)
            acc = wmma_bf(frag_a_bf(Xc, DD, m0, k0), frag_bt_bf(W, DD, n0, k0), acc);
        st_c_bf16(uv ? Cv : Cu, DD, m0, n0, acc);
    }
    for (int i = tid; i < CC * CC; i += nthr) Ps[i] = 0.0f;
    __syncthreads();

    for (int h = 0; h < HH; ++h) {
        int kh = h << 4;
        for (int job = wave; job < 8 * 8; job += nw) {
            int m0 = (job >> 3) << 4, n0 = (job & 7) << 4;
            v8f acc = {};
            acc = wmma_bf(frag_a_bf_k16(Cu, DD, m0, kh), frag_bt_bf_k16(Cv, DD, n0, kh), acc);
            st_c_f32(Sc, CC, m0, n0, acc, 0.25f);
        }
        __syncthreads();
        for (int r = tid; r < CC; r += nthr) {
            const float* row = Sc + r * CC;
            float mx = -3.4e38f;
            for (int qq = 0; qq < CC; ++qq) mx = fmaxf(mx, row[qq]);
            float sum = 0.0f;
            for (int qq = 0; qq < CC; ++qq) sum += __expf(row[qq] - mx);
            float inv = 1.0f / (sum * (float)HH);
            float* prow = Ps + r * CC;
            for (int qq = 0; qq < CC; ++qq) prow[qq] += __expf(row[qq] - mx) * inv;
        }
        __syncthreads();
    }

    // msg += mean_attn * qzn  (elementwise; C == D)
    for (int i = tid; i < CC * DD; i += nthr) {
        int cr = i >> 7, d = i & 127;
        size_t g = (((size_t)b * CC + cr) * PP + p) * DD + d;
        msg[g] += Ps[i] * qzn[g];
    }
}

// ---------------------------------------------------------------------------
// Kernel 4: topic messages, 64 rows per WG.  msg += mg
// LDS: Tb 128K | Xr 16K | Qg 128K | rsc = 272.25 KB
// ---------------------------------------------------------------------------
#define SM_TOPIC (131072 + 16384 + 131072 + 256)
__global__ void k_topic(const unsigned short* __restrict__ qznb,
                        const unsigned short* __restrict__ Tg, float* __restrict__ msg) {
    extern __shared__ char smem[];
    unsigned short* Tb = (unsigned short*)smem;                    // [512][128]
    unsigned short* Xr = (unsigned short*)(smem + 131072);         // [64][128]
    float* Qg  = (float*)(smem + 131072 + 16384);                  // [64][512]
    float* rsc = (float*)(smem + 131072 + 16384 + 131072);         // [64]

    int b  = blockIdx.x / 192;
    int r0 = (blockIdx.x % 192) * 64;
    int tid = threadIdx.x, nthr = blockDim.x;
    int wave = tid >> 5, nw = nthr >> 5;

    if (wave == 0) {
        tdm_load_bf16_2d(qznb + ((size_t)b * (CC * PP) + r0) * DD, Xr, 64 * DD, 1, 64 * DD);
        tdm_load_bf16_2d(Tg + (size_t)b * GG * DD, Tb, DD, GG, DD);  // 65536 elems as 2-D
        tdm_wait0();
    }
    __syncthreads();

    // Qg = relu(X @ T^T)   [64 x 512]
    for (int job = wave; job < 4 * 32; job += nw) {
        int m0 = (job >> 5) << 4, n0 = (job & 31) << 4;
        v8f acc = {};
#pragma unroll
        for (int k0 = 0; k0 < DD; k0 += 32)
            acc = wmma_bf(frag_a_bf(Xr, DD, m0, k0), frag_bt_bf(Tb, DD, n0, k0), acc);
        st_c_f32_relu(Qg, GG, m0, n0, acc);
    }
    __syncthreads();
    for (int r = tid; r < 64; r += nthr) {
        const float* row = Qg + r * GG;
        float s = 0.0f;
        for (int g = 0; g < GG; ++g) s += row[g];
        rsc[r] = 1.0f / fmaxf(s, 1e-6f);
    }
    __syncthreads();

    // mg = (Qg * rsc) @ T   (sum over g = 512)
    float* Mb = msg + ((size_t)b * (CC * PP) + r0) * DD;
    for (int job = wave; job < 4 * 8; job += nw) {
        int m0 = (job >> 3) << 4, n0 = (job & 7) << 4;
        v8f acc = {};
        for (int k0 = 0; k0 < GG; k0 += 32)
            acc = wmma_bf(frag_a_f32(Qg, GG, m0, k0, rsc), frag_b_bf(Tb, DD, k0, n0), acc);
        int lane = threadIdx.x & 31, N = lane & 15, half = lane >> 4;
#pragma unroll
        for (int i = 0; i < 8; ++i) {
            size_t idx = (size_t)(m0 + i + (half << 3)) * DD + n0 + N;
            Mb[idx] += acc[i];
        }
    }
}

// ---------------------------------------------------------------------------
// Kernel 5: qz2 = 0.5*(unary + msg + qz_old) -> d_out
// ---------------------------------------------------------------------------
__global__ void k_combine(const float4* __restrict__ qz, const float4* __restrict__ un,
                          const float4* __restrict__ m, float4* __restrict__ out, int n4) {
    int i = blockIdx.x * blockDim.x + threadIdx.x;
    if (i >= n4) return;
    float4 a = qz[i], u = un[i], s = m[i], o;
    o.x = 0.5f * (u.x + s.x + a.x);
    o.y = 0.5f * (u.y + s.y + a.y);
    o.z = 0.5f * (u.z + s.z + a.z);
    o.w = 0.5f * (u.w + s.w + a.w);
    out[i] = o;
}

// ---------------------------------------------------------------------------
// Kernel 6: MLP + residual, in-place on d_out.  64 rows per WG.
// ---------------------------------------------------------------------------
#define SM_MLP (32768 + 32768 + 16384 + 16384 + 512 * 4)
__global__ void k_mlp(float* __restrict__ out, const float* __restrict__ gw,
                      const float* __restrict__ bw, const unsigned short* __restrict__ w1b,
                      const float* __restrict__ b1, const unsigned short* __restrict__ w2b,
                      const float* __restrict__ b2) {
    extern __shared__ char smem[];
    unsigned short* W1 = (unsigned short*)smem;                    // [128][128]
    unsigned short* W2 = (unsigned short*)(smem + 32768);
    unsigned short* Xh = (unsigned short*)(smem + 65536);          // [64][128]
    unsigned short* Hm = (unsigned short*)(smem + 65536 + 16384);
    float* B1 = (float*)(smem + 65536 + 32768);
    float* B2 = (float*)(smem + 65536 + 32768 + 512);
    float* Gm = (float*)(smem + 65536 + 32768 + 1024);
    float* Bt = (float*)(smem + 65536 + 32768 + 1536);

    size_t r0 = (size_t)blockIdx.x * 64;
    int tid = threadIdx.x, nthr = blockDim.x;
    int wave = tid >> 5, nw = nthr >> 5;

    if (wave == 0) {
        tdm_load_bf16_2d(w1b, W1, DD * DD, 1, DD * DD);
        tdm_load_bf16_2d(w2b, W2, DD * DD, 1, DD * DD);
    }
    for (int i = tid; i < DD; i += nthr) { B1[i] = b1[i]; B2[i] = b2[i]; Gm[i] = gw[i]; Bt[i] = bw[i]; }
    __syncthreads();

    // layernorm rows -> Xh (bf16)
    for (int r = tid; r < 64; r += nthr) {
        const float* row = out + (r0 + r) * DD;
        float mu = 0.0f;
        for (int d = 0; d < DD; ++d) mu += row[d];
        mu *= (1.0f / 128.0f);
        float var = 0.0f;
        for (int d = 0; d < DD; ++d) { float dx = row[d] - mu; var += dx * dx; }
        float rstd = rsqrtf(var * (1.0f / 128.0f) + 1e-5f);
        for (int d = 0; d < DD; ++d)
            Xh[r * DD + d] = f2bf((row[d] - mu) * rstd * Gm[d] + Bt[d]);
    }
    if (wave == 0) tdm_wait0();   // W1/W2 resident before matmuls
    __syncthreads();

    // Hm = gelu(Xh @ W1^T + b1)
    for (int job = wave; job < 4 * 8; job += nw) {
        int m0 = (job >> 3) << 4, n0 = (job & 7) << 4;
        v8f acc = {};
#pragma unroll
        for (int k0 = 0; k0 < DD; k0 += 32)
            acc = wmma_bf(frag_a_bf(Xh, DD, m0, k0), frag_bt_bf(W1, DD, n0, k0), acc);
        int lane = threadIdx.x & 31, N = lane & 15, half = lane >> 4;
#pragma unroll
        for (int i = 0; i < 8; ++i) {
            float v = acc[i] + B1[n0 + N];
            v = 0.5f * v * (1.0f + erff(v * 0.70710678118654752f));  // exact gelu
            Hm[(m0 + i + (half << 3)) * DD + n0 + N] = f2bf(v);
        }
    }
    __syncthreads();

    // out += Hm @ W2^T + b2
    for (int job = wave; job < 4 * 8; job += nw) {
        int m0 = (job >> 3) << 4, n0 = (job & 7) << 4;
        v8f acc = {};
#pragma unroll
        for (int k0 = 0; k0 < DD; k0 += 32)
            acc = wmma_bf(frag_a_bf(Hm, DD, m0, k0), frag_bt_bf(W2, DD, n0, k0), acc);
        int lane = threadIdx.x & 31, N = lane & 15, half = lane >> 4;
#pragma unroll
        for (int i = 0; i < 8; ++i) {
            size_t idx = (r0 + m0 + i + (half << 3)) * DD + n0 + N;
            out[idx] = out[idx] + acc[i] + B2[n0 + N];
        }
    }
}

// ---------------------------------------------------------------------------
// Launch
// ---------------------------------------------------------------------------
extern "C" void kernel_launch(void* const* d_in, const int* in_sizes, int n_in,
                              void* d_out, int out_size, void* d_ws, size_t ws_size,
                              hipStream_t stream) {
    const float* qz       = (const float*)d_in[0];
    const float* unary    = (const float*)d_in[1];
    const float* time_u   = (const float*)d_in[2];
    const float* time_v   = (const float*)d_in[3];
    const float* chan_u   = (const float*)d_in[4];
    const float* chan_v   = (const float*)d_in[5];
    const float* topic    = (const float*)d_in[6];
    const float* ln_g     = (const float*)d_in[7];
    const float* ln_b     = (const float*)d_in[8];
    const float* w1       = (const float*)d_in[9];
    const float* b1       = (const float*)d_in[10];
    const float* w2       = (const float*)d_in[11];
    const float* w2bias   = (const float*)d_in[12];
    float* out = (float*)d_out;

    // workspace layout
    float* qzn = (float*)d_ws;                         // NELEM f32
    float* msg = qzn + NELEM;                          // NELEM f32
    unsigned short* qznb = (unsigned short*)(msg + NELEM);   // NELEM bf16
    unsigned short* tub  = qznb + NELEM;               // B*D*D each
    unsigned short* tvb  = tub + BB * DD * DD;
    unsigned short* cub  = tvb + BB * DD * DD;
    unsigned short* cvb  = cub + BB * DD * DD;
    unsigned short* tpb  = cvb + BB * DD * DD;         // B*G*D
    unsigned short* w1b  = tpb + (size_t)BB * GG * DD;
    unsigned short* w2b  = w1b + DD * DD;

    // 0) weight prepass: f32 -> bf16
    k_f32_to_bf16<<<256, 256, 0, stream>>>(time_u, tub, BB * DD * DD);
    k_f32_to_bf16<<<256, 256, 0, stream>>>(time_v, tvb, BB * DD * DD);
    k_f32_to_bf16<<<256, 256, 0, stream>>>(chan_u, cub, BB * DD * DD);
    k_f32_to_bf16<<<256, 256, 0, stream>>>(chan_v, cvb, BB * DD * DD);
    k_f32_to_bf16<<<512, 256, 0, stream>>>(topic, tpb, BB * GG * DD);
    k_f32_to_bf16<<<64, 256, 0, stream>>>(w1, w1b, DD * DD);
    k_f32_to_bf16<<<64, 256, 0, stream>>>(w2, w2b, DD * DD);

    // 1) qzn = LN(qz), f32 + bf16 mirror
    k_layernorm<<<NROWS / 8, 256, 0, stream>>>(qz, ln_g, ln_b, qzn, qznb, NROWS);
    // 2) msg = time messages
    k_time<<<BB * CC, 128, SM_TIME, stream>>>(qzn, qznb, tub, tvb, msg);
    // 3) msg += channel messages
    k_channel<<<BB * PP, 128, SM_CH, stream>>>(qzn, qznb, cub, cvb, msg);
    // 4) msg += topic messages
    k_topic<<<BB * 192, 128, SM_TOPIC, stream>>>(qznb, tpb, msg);
    // 5) out = 0.5*(unary + msg + qz)
    {
        int n4 = (int)(NELEM / 4);
        k_combine<<<(n4 + 255) / 256, 256, 0, stream>>>(
            (const float4*)qz, (const float4*)unary, (const float4*)msg, (float4*)out, n4);
    }
    // 6) out += MLP(LN(out))
    k_mlp<<<NROWS / 64, 128, SM_MLP, stream>>>(out, ln_g, ln_b, w1b, b1, w2b, w2bias);
}
